// MutualDTA_78065325572488
// MI455X (gfx1250) — matrix operand order repaired and checked by reading, not compiled, not run on hardware
//
#include <hip/hip_runtime.h>
#include <hip/hip_bf16.h>
#include <math.h>

// ---------------------------------------------------------------------------
// MutualDTA forward pass for gfx1250 (MI455X).
// All dense GEMMs run on v_wmma_f32_16x16x32_bf16 (bf16 A/B staged in
// workspace, fp32 accumulate). Sequential Mamba scan parallelized over
// (batch, head, headdim) with LDS-shared B/C. Segment ops via f32 atomics.
// Input order assumption: setup_inputs() dict insertion order with the
// nested params dict expanded in its insertion order at its position.
// ---------------------------------------------------------------------------

#define CBG   16
#define CNPG  64
#define CEPG  128
#define GN    1024      // nodes
#define GE    2048      // edges
#define GELG  6144      // line-graph edges
#define SEQL  512
#define CDX   512
#define CDP   512
#define CDE   6
#define CDH   128
#define CPH   256
#define CDF   64
#define CKA   64
#define CDIN  512
#define CDST  64
#define CNH   8
#define CHD   64
#define CDXBC 640
#define CZ    1160      // 2*DIN + 2*DST + NH

typedef __attribute__((ext_vector_type(16))) __bf16 v16bf;
typedef __attribute__((ext_vector_type(8)))  __bf16 v8bf;
typedef __attribute__((ext_vector_type(8)))  float  v8f;

// ---------------------------------------------------------------------------
// Generic WMMA GEMM:  C[M,N] = act( A[M,K](bf16) * B[N,K]^T(bf16) + bias + add )
// One 16x16 tile per wave, 4 waves per block along N. Batched via strides.
// act: 0 = none, 1 = leaky(0.01), 2 = tanh
// ---------------------------------------------------------------------------
__global__ __launch_bounds__(128) void k_gemm_wmma(
    const __bf16* __restrict__ A, const __bf16* __restrict__ B,
    float* __restrict__ C, const float* __restrict__ bias,
    const float* __restrict__ add, int M, int N, int K, int act,
    long long sA, long long sB, long long sC, long long sAdd)
{
    const int lane = threadIdx.x & 31;
    const int wave = threadIdx.x >> 5;
    const int n0 = blockIdx.x * 64 + wave * 16;
    const int m0 = blockIdx.y * 16;
    const int bz = blockIdx.z;

    A += (long long)bz * sA;
    B += (long long)bz * sB;
    C += (long long)bz * sC;
    const float* addp = add ? add + (long long)bz * sAdd : nullptr;

    const int half = lane >> 4;
    const int l15  = lane & 15;
    int ra = m0 + l15; if (ra >= M) ra = M - 1;   // clamp; stores are guarded
    int rb = n0 + l15; if (rb >= N) rb = N - 1;
    const __bf16* Ap = A + (long long)ra * K;
    const __bf16* Bp = B + (long long)rb * K;

    v8f acc = {0.f, 0.f, 0.f, 0.f, 0.f, 0.f, 0.f, 0.f};

    for (int k0 = 0; k0 < K; k0 += 32) {
        __builtin_prefetch(Ap + k0 + 256, 0, 0);   // global_prefetch_b8
        __builtin_prefetch(Bp + k0 + 256, 0, 0);
        // A 16x32 bf16 fragment: lane(l<16) row m0+l holds K={0..7,16..23}(+8*half)
        v8bf a0 = *(const v8bf*)(Ap + k0 + 8 * half);
        v8bf a1 = *(const v8bf*)(Ap + k0 + 16 + 8 * half);
        // B 32x16 fragment (B stored [N,K]): lane col n0+(l&15), K={0..7,16..23}(+8*half)
        v8bf b0 = *(const v8bf*)(Bp + k0 + 8 * half);
        v8bf b1 = *(const v8bf*)(Bp + k0 + 16 + 8 * half);
        v16bf av, bv;
#pragma unroll
        for (int i = 0; i < 8; ++i) {
            av[i] = a0[i]; av[i + 8] = a1[i];
            bv[i] = b0[i]; bv[i + 8] = b1[i];
        }
        acc = __builtin_amdgcn_wmma_f32_16x16x32_bf16(
            false, av, false, bv, (short)0, acc, false, false);
    }

    const int n = n0 + l15;
    if (n < N) {
        const float bsv = bias ? bias[n] : 0.f;
#pragma unroll
        for (int r = 0; r < 8; ++r) {
            int m = m0 + r + 8 * half;           // C/D layout: M = r + 8*half
            if (m < M) {
                float v = acc[r] + bsv;
                if (addp) v += addp[(long long)m * N + n];
                if (act == 1)      v = v > 0.f ? v : 0.01f * v;
                else if (act == 2) v = tanhf(v);
                C[(long long)m * N + n] = v;
            }
        }
    }
}

// ----------------------------- utility kernels -----------------------------
__global__ void k_fill(float* p, float v, long long n) {
    long long i = (long long)blockIdx.x * blockDim.x + threadIdx.x;
    if (i < n) p[i] = v;
}
__global__ void k_copy(const float* s, float* d, long long n) {
    long long i = (long long)blockIdx.x * blockDim.x + threadIdx.x;
    if (i < n) d[i] = s[i];
}
__global__ void k_add2(const float* a, const float* b, float* d, long long n) {
    long long i = (long long)blockIdx.x * blockDim.x + threadIdx.x;
    if (i < n) d[i] = a[i] + b[i];
}
__global__ void k_cvt(const float* s, __bf16* d, long long n) {
    long long i = (long long)blockIdx.x * blockDim.x + threadIdx.x;
    if (i < n) d[i] = (__bf16)s[i];
}
// f32 [R,C] -> bf16 [C,R]  (weight transpose+convert for B-fragments)
__global__ void k_cvt_t(const float* s, __bf16* d, int R, int C) {
    long long i = (long long)blockIdx.x * blockDim.x + threadIdx.x;
    if (i < (long long)R * C) {
        int r = (int)(i / C), c = (int)(i % C);
        d[(long long)c * R + r] = (__bf16)s[i];
    }
}

// ------------------------------ graph kernels ------------------------------
__global__ void k_edge_init(const float* __restrict__ eu, const float* __restrict__ ev,
                            const float* __restrict__ ea6, const float* __restrict__ We,
                            const int* __restrict__ ei, float* __restrict__ eattr)
{
    int e = blockIdx.x, f = threadIdx.x;      // grid GE, block CDF
    int src = ei[e], dst = ei[GE + e];
    float v = 0.f;
#pragma unroll
    for (int k = 0; k < CDE; ++k) v += ea6[e * CDE + k] * We[k * CDF + f];
    eattr[(long long)e * CDF + f] =
        (eu[(long long)src * CDF + f] + ev[(long long)dst * CDF + f] + v) * (1.f / 3.f);
}

// scatter-add of CDF-vectors: acc[dst[e]] += src[srcIdx ? srcIdx[e] : e]
__global__ void k_scatter_add(const float* __restrict__ src, const int* __restrict__ idx_src,
                              const int* __restrict__ idx_dst, float* acc, long long nedge)
{
    long long i = (long long)blockIdx.x * blockDim.x + threadIdx.x;
    if (i >= nedge * CDF) return;
    long long e = i / CDF; int f = (int)(i % CDF);
    long long se = idx_src ? (long long)idx_src[e] : e;
    atomicAdd(&acc[(long long)idx_dst[e] * CDF + f], src[se * CDF + f]);
}

__global__ void k_xconv(const float* __restrict__ s2, const float* __restrict__ out,
                        const float* __restrict__ Wrel, const float* __restrict__ brel,
                        const float* __restrict__ Wroot, float* xc)
{
    int e = blockIdx.x * 256 + threadIdx.x;
    if (e >= GE) return;
    float v = brel[0];
    for (int f = 0; f < CDF; ++f)
        v += s2[(long long)e * CDF + f] * Wrel[f] + out[(long long)e * CDF + f] * Wroot[f];
    xc[e] = v;
}

__global__ void k_segsoftmax(const float* __restrict__ xc, float* sc) {
    __shared__ float red[CEPG];
    int g = blockIdx.x, t = threadIdx.x;        // block CEPG=128
    float v = xc[g * CEPG + t];
    red[t] = v; __syncthreads();
    for (int s = CEPG / 2; s > 0; s >>= 1) { if (t < s) red[t] = fmaxf(red[t], red[t + s]); __syncthreads(); }
    float m = red[0]; __syncthreads();
    float e = expf(v - m); red[t] = e; __syncthreads();
    for (int s = CEPG / 2; s > 0; s >>= 1) { if (t < s) red[t] += red[t + s]; __syncthreads(); }
    sc[g * CEPG + t] = e / red[0];
}

__global__ void k_gx(const float* __restrict__ out, const float* __restrict__ sc, float* gx) {
    int g = blockIdx.x, f = threadIdx.x;        // grid CBG, block CDF
    float v = 0.f;
    for (int e = 0; e < CEPG; ++e) {
        int ee = g * CEPG + e;
        v += out[(long long)ee * CDF + f] * sc[ee];
    }
    gx[g * CDF + f] = v;
}

__global__ void k_scores(const float* g0, const float* g1, const float* g2,
                         const float* watt, float* sco)
{
    __shared__ float red[CDF]; __shared__ float lg[3];
    int b = blockIdx.x, f = threadIdx.x;        // grid CBG, block CDF
    const float* gs[3] = {g0, g1, g2};
    for (int t = 0; t < 3; ++t) {
        red[f] = gs[t][b * CDF + f] * watt[f]; __syncthreads();
        for (int s = CDF / 2; s > 0; s >>= 1) { if (f < s) red[f] += red[f + s]; __syncthreads(); }
        if (f == 0) lg[t] = red[0] * 0.125f;    // DF^-0.5 = 64^-0.5
        __syncthreads();
    }
    if (f == 0) {
        float m = fmaxf(lg[0], fmaxf(lg[1], lg[2]));
        float e0 = expf(lg[0] - m), e1 = expf(lg[1] - m), e2 = expf(lg[2] - m);
        float s = e0 + e1 + e2;
        sco[b * 3 + 0] = e0 / s; sco[b * 3 + 1] = e1 / s; sco[b * 3 + 2] = e2 / s;
    }
}

__global__ void k_mixout(const float* o0, const float* o1, const float* o2,
                         const float* sco, const int* eb, float* o)
{
    long long i = (long long)blockIdx.x * blockDim.x + threadIdx.x;
    if (i >= (long long)GE * CDF) return;
    int e = (int)(i / CDF), b = eb[e];
    o[i] = o0[i] * sco[b * 3] + o1[i] * sco[b * 3 + 1] + o2[i] * sco[b * 3 + 2];
}

__global__ void k_denc(const float* enc, const float* xdp, float* denc, __bf16* dbf) {
    long long i = (long long)blockIdx.x * blockDim.x + threadIdx.x;
    if (i >= (long long)GN * CDH) return;
    float v = 0.01f * enc[i] + 0.99f * xdp[i];
    denc[i] = v; dbf[i] = (__bf16)v;
}

// ------------------------------ mamba kernels ------------------------------
__global__ void k_dtda(const float* __restrict__ zx, const float* dtb, const float* Alog,
                       float* dt, float* dA)
{
    long long i = (long long)blockIdx.x * blockDim.x + threadIdx.x;
    if (i >= (long long)CBG * SEQL * CNH) return;
    int h = (int)(i % CNH); long long row = i / CNH;
    float dtr = zx[row * CZ + CDIN + CDXBC + h] + dtb[h];
    float sp = dtr > 20.f ? dtr : log1pf(expf(dtr));
    dt[i] = sp;
    dA[i] = expf(sp * -expf(Alog[h]));
}

__global__ void k_conv(const float* __restrict__ zx, const float* __restrict__ cw,
                       const float* __restrict__ cb, float* xBC)
{
    long long i = (long long)blockIdx.x * blockDim.x + threadIdx.x;
    if (i >= (long long)CBG * SEQL * CDXBC) return;
    int c = (int)(i % CDXBC); long long bl = i / CDXBC;
    int l = (int)(bl % SEQL); long long b = bl / SEQL;
    float v = cb[c];
#pragma unroll
    for (int k = 0; k < 4; ++k) {
        int ls = l + k - 3;
        if (ls >= 0) v += zx[(b * SEQL + ls) * CZ + CDIN + c] * cw[c * 4 + k];
    }
    xBC[i] = v / (1.f + expf(-v));   // silu
}

// selective scan: one block per (b,h); thread = p (headdim); h-state in regs.
__global__ __launch_bounds__(64) void k_scan(const float* __restrict__ xBC,
                                             const float* __restrict__ dA,
                                             const float* __restrict__ dt,
                                             float* __restrict__ ys)
{
    __shared__ float sB[CDST], sC[CDST];
    const int p = threadIdx.x;
    const int h = blockIdx.x % CNH;
    const long long b = blockIdx.x / CNH;
    float hs[CDST];
#pragma unroll
    for (int n = 0; n < CDST; ++n) hs[n] = 0.f;
    for (int t = 0; t < SEQL; ++t) {
        long long row = b * SEQL + t;
        sB[p] = xBC[row * CDXBC + CDIN + p];
        sC[p] = xBC[row * CDXBC + CDIN + CDST + p];
        __syncthreads();
        float dAt = dA[row * CNH + h];
        float dtt = dt[row * CNH + h];
        float xt  = xBC[row * CDXBC + h * CHD + p];
        float coef = dtt * xt, acc = 0.f;
#pragma unroll
        for (int n = 0; n < CDST; ++n) { hs[n] = hs[n] * dAt + coef * sB[n]; acc += hs[n] * sC[n]; }
        ys[row * CDIN + h * CHD + p] = acc;
        __syncthreads();
    }
}

// y = (ys + D*x) * silu(z); RMS-norm over DIN; emit bf16 for m_out GEMM
__global__ __launch_bounds__(256) void k_ypost(const float* __restrict__ ys,
                                               const float* __restrict__ xBC,
                                               const float* __restrict__ zx,
                                               const float* mD, const float* nw,
                                               __bf16* __restrict__ ybf)
{
    __shared__ float red[256];
    long long r = blockIdx.x; int t = threadIdx.x;
    float vals[2]; float ss = 0.f;
#pragma unroll
    for (int j = 0; j < 2; ++j) {
        int d = t + j * 256;
        float v = ys[r * CDIN + d] + mD[d >> 6] * xBC[r * CDXBC + d];
        float z = zx[r * CZ + d];
        v *= z / (1.f + expf(-z));
        vals[j] = v; ss += v * v;
    }
    red[t] = ss; __syncthreads();
    for (int s = 128; s > 0; s >>= 1) { if (t < s) red[t] += red[t + s]; __syncthreads(); }
    float rms = rsqrtf(red[0] / (float)CDIN + 1e-5f);
#pragma unroll
    for (int j = 0; j < 2; ++j) {
        int d = t + j * 256;
        ybf[r * CDIN + d] = (__bf16)(vals[j] * rms * nw[d]);
    }
}

__global__ void k_tmmean(const float* mo, const float* tg, float* o) {
    long long i = (long long)blockIdx.x * blockDim.x + threadIdx.x;
    if (i >= (long long)CBG * CPH) return;
    int b = (int)(i / CPH), d = (int)(i % CPH);
    float s = 0.f;
    for (int l = 0; l < SEQL; ++l) {
        long long idx = ((long long)b * SEQL + l) * CPH + d;
        s += 0.5f * mo[idx] + 0.5f * tg[idx];
    }
    o[i] = s * (1.f / SEQL);
}

// ---------------------- bilinear-attention tail kernels --------------------
// a[b,t] = softmax_t( sum_k w[k]*H[b,k,t] ), block = Ncols threads (dyn LDS)
__global__ void k_attn(const float* __restrict__ H, const float* __restrict__ w,
                       float* __restrict__ a, int Ncols)
{
    extern __shared__ float red[];
    int b = blockIdx.x, t = threadIdx.x;
    float v = 0.f;
    for (int k = 0; k < CKA; ++k) v += w[k] * H[((long long)b * CKA + k) * Ncols + t];
    red[t] = v; __syncthreads();
    for (int s = blockDim.x >> 1; s > 0; s >>= 1) { if (t < s) red[t] = fmaxf(red[t], red[t + s]); __syncthreads(); }
    float m = red[0]; __syncthreads();
    float e = expf(v - m); red[t] = e; __syncthreads();
    for (int s = blockDim.x >> 1; s > 0; s >>= 1) { if (t < s) red[t] += red[t + s]; __syncthreads(); }
    a[b * Ncols + t] = e / red[0];
}

// o[b,d] = sum_r a[b,r]*X[b,r,d], block = D threads
__global__ void k_wsum(const float* __restrict__ a, const float* __restrict__ X,
                       float* o, int R, int D)
{
    int b = blockIdx.x, d = threadIdx.x;
    float v = 0.f;
    for (int r = 0; r < R; ++r) v += a[b * R + r] * X[((long long)b * R + r) * D + d];
    o[b * D + d] = v;
}

__global__ void k_xmean(const float* x, float* o) {
    int b = blockIdx.x, d = threadIdx.x;      // grid CBG, block CDH
    float v = 0.f;
    for (int i = 0; i < CNPG; ++i) v += x[((long long)b * CNPG + i) * CDH + d];
    o[b * CDH + d] = v * (1.f / CNPG);
}

__global__ void k_feat(const float* xm, const float* cv, const float* tmm,
                       const float* pv, float* feat)
{
    int i = blockIdx.x * 256 + threadIdx.x;
    if (i >= CBG * 768) return;
    int b = i / 768, d = i % 768;
    float v;
    if (d < 128)      v = xm[b * 128 + d];
    else if (d < 256) v = cv[b * 128 + (d - 128)];
    else if (d < 512) v = tmm[b * 256 + (d - 256)];
    else              v = pv[b * 256 + (d - 512)];
    feat[i] = v;
}

__global__ __launch_bounds__(256) void k_ln(const float* __restrict__ x, const float* g,
                                            const float* bt, __bf16* out, int D)
{
    __shared__ float r1[256], r2[256];
    int row = blockIdx.x, t = threadIdx.x;
    float s = 0.f, s2 = 0.f;
    for (int d = t; d < D; d += 256) { float v = x[(long long)row * D + d]; s += v; s2 += v * v; }
    r1[t] = s; r2[t] = s2; __syncthreads();
    for (int st = 128; st > 0; st >>= 1) {
        if (t < st) { r1[t] += r1[t + st]; r2[t] += r2[t + st]; } __syncthreads();
    }
    float mu = r1[0] / D;
    float inv = rsqrtf(r2[0] / D - mu * mu + 1e-5f);
    for (int d = t; d < D; d += 256)
        out[(long long)row * D + d] = (__bf16)((x[(long long)row * D + d] - mu) * inv * g[d] + bt[d]);
}

__global__ void k_final(const float* h2, const float* W3, const float* b3, float* out) {
    int b = threadIdx.x;
    if (b >= CBG) return;
    float v = b3[0];
    for (int d = 0; d < CDH; ++d) v += h2[b * CDH + d] * W3[d];
    out[b] = v;
}

// ---------------------------------------------------------------------------
extern "C" void kernel_launch(void* const* d_in, const int* in_sizes, int n_in,
                              void* d_out, int out_size, void* d_ws, size_t ws_size,
                              hipStream_t stream)
{
    (void)in_sizes; (void)n_in; (void)out_size; (void)ws_size;
    // ---- unpack inputs (setup_inputs order, params expanded in place) ----
    const float* node_x    = (const float*)d_in[0];
    const float* edge_attr = (const float*)d_in[1];
    const float* proteins  = (const float*)d_in[2];
    int pi = 3;
    const float* W0      = (const float*)d_in[pi++];
    const float* b0      = (const float*)d_in[pi++];
    const float* dm_W    = (const float*)d_in[pi++];
    const float* dm_b    = (const float*)d_in[pi++];
    const float* Wu      = (const float*)d_in[pi++];
    const float* Wv      = (const float*)d_in[pi++];
    const float* We      = (const float*)d_in[pi++];
    const float* attWrel = (const float*)d_in[pi++];
    const float* attbrel = (const float*)d_in[pi++];
    const float* attWroot= (const float*)d_in[pi++];
    const float* Wg      = (const float*)d_in[pi++];
    const float* bg      = (const float*)d_in[pi++];
    const float* w_att   = (const float*)d_in[pi++];
    const float* Wa      = (const float*)d_in[pi++];
    const float* ba      = (const float*)d_in[pi++];
    const float* Wp0     = (const float*)d_in[pi++];
    const float* bp0     = (const float*)d_in[pi++];
    const float* W_b     = (const float*)d_in[pi++];
    const float* W_x     = (const float*)d_in[pi++];
    const float* W_p     = (const float*)d_in[pi++];
    const float* w_hx    = (const float*)d_in[pi++];
    const float* w_hp    = (const float*)d_in[pi++];
    const float* m_in_W  = (const float*)d_in[pi++];
    const float* m_conv_w= (const float*)d_in[pi++];
    const float* m_conv_b= (const float*)d_in[pi++];
    const float* m_dt_b  = (const float*)d_in[pi++];
    const float* m_A_log = (const float*)d_in[pi++];
    const float* m_D     = (const float*)d_in[pi++];
    const float* m_norm_w= (const float*)d_in[pi++];
    const float* m_out_W = (const float*)d_in[pi++];
    const float* ln1_g   = (const float*)d_in[pi++];
    const float* ln1_b   = (const float*)d_in[pi++];
    const float* W1      = (const float*)d_in[pi++];
    const float* b1      = (const float*)d_in[pi++];
    const float* ln2_g   = (const float*)d_in[pi++];
    const float* ln2_b   = (const float*)d_in[pi++];
    const float* W2      = (const float*)d_in[pi++];
    const float* b2      = (const float*)d_in[pi++];
    const float* W3      = (const float*)d_in[pi++];
    const float* b3      = (const float*)d_in[pi++];
    const int*   eidx    = (const int*)d_in[pi++];
    const int*   lgidx   = (const int*)d_in[pi++];
    const int*   ebatch  = (const int*)d_in[pi++];

    // ---- workspace allocator ----
    char* base = (char*)d_ws;
    size_t off = 0;
    auto alloc = [&](size_t bytes) -> void* {
        void* p = base + off;
        off += (bytes + 255) & ~(size_t)255;
        return p;
    };
    auto fbuf = [&](size_t n) { return (float*)alloc(n * sizeof(float)); };
    auto hbuf = [&](size_t n) { return (__bf16*)alloc(n * sizeof(__bf16)); };

    // bf16 weights (transposed to [N,K]) and activations
    __bf16* nodex_bf = hbuf((size_t)GN * CDX);
    __bf16* prot_bf  = hbuf((size_t)CBG * SEQL * CDP);
    __bf16* W0T   = hbuf((size_t)CDH * CDX);
    __bf16* dmT   = hbuf((size_t)CDF * CDH);
    __bf16* WuT   = hbuf((size_t)CDF * CDF);
    __bf16* WvT   = hbuf((size_t)CDF * CDF);
    __bf16* WgT   = hbuf((size_t)CDF * CDF);
    __bf16* WaT   = hbuf((size_t)CDH * CDF);
    __bf16* Wp0T  = hbuf((size_t)CPH * CDP);
    __bf16* minT  = hbuf((size_t)CZ * CPH);
    __bf16* moutT = hbuf((size_t)CPH * CDIN);
    __bf16* WbT   = hbuf((size_t)CDH * CPH);
    __bf16* Wx_bf = hbuf((size_t)CKA * CDH);
    __bf16* Wp_bf = hbuf((size_t)CKA * CPH);
    __bf16* W1T   = hbuf((size_t)768 * 768);
    __bf16* W2T   = hbuf((size_t)CDH * 768);

    float*  xd_pre = fbuf((size_t)GN * CDH);   __bf16* xdp_bf = hbuf((size_t)GN * CDH);
    float*  x_in   = fbuf((size_t)GN * CDF);   __bf16* xin_bf = hbuf((size_t)GN * CDF);
    float*  eu     = fbuf((size_t)GN * CDF);
    float*  ev     = fbuf((size_t)GN * CDF);
    float*  e_attr = fbuf((size_t)GE * CDF);
    float*  outs[3]; for (int t = 0; t < 3; ++t) outs[t] = fbuf((size_t)GE * CDF);
    float*  segA   = fbuf((size_t)GE * CDF);
    float*  segB   = fbuf((size_t)GE * CDF);
    float*  xconv  = fbuf(GE);
    float*  scv    = fbuf(GE);
    float*  gx     = fbuf((size_t)CBG * CDF);  __bf16* gx_bf = hbuf((size_t)CBG * CDF);
    float*  gout[3]; for (int t = 0; t < 3; ++t) gout[t] = fbuf((size_t)CBG * CDF);
    float*  scores = fbuf((size_t)CBG * 3);
    float*  outF   = fbuf((size_t)GE * CDF);
    float*  dn     = fbuf((size_t)GN * CDF);   __bf16* dn_bf = hbuf((size_t)GN * CDF);
    float*  encT   = fbuf((size_t)GN * CDH);
    float*  denc   = fbuf((size_t)GN * CDH);   __bf16* dd_bf = hbuf((size_t)GN * CDH);
    float*  target = fbuf((size_t)CBG * SEQL * CPH);
    __bf16* tgt_bf = hbuf((size_t)CBG * SEQL * CPH);
    float*  zx     = fbuf((size_t)CBG * SEQL * CZ);
    float*  dtb_   = fbuf((size_t)CBG * SEQL * CNH);
    float*  dAb    = fbuf((size_t)CBG * SEQL * CNH);
    float*  xBC    = fbuf((size_t)CBG * SEQL * CDXBC);
    float*  ys     = fbuf((size_t)CBG * SEQL * CDIN);
    __bf16* y_bf   = hbuf((size_t)CBG * SEQL * CDIN);
    float*  mo     = fbuf((size_t)CBG * SEQL * CPH);
    float*  tmm    = fbuf((size_t)CBG * CPH);
    float*  T2     = fbuf((size_t)CBG * SEQL * CDH); __bf16* T2_bf = hbuf((size_t)CBG * SEQL * CDH);
    float*  Cc     = fbuf((size_t)CBG * SEQL * CDF); __bf16* Cc_bf = hbuf((size_t)CBG * SEQL * CDF);
    float*  CcT    = fbuf((size_t)CBG * CDF * SEQL); __bf16* CcT_bf= hbuf((size_t)CBG * CDF * SEQL);
    float*  WxX    = fbuf((size_t)CBG * CKA * CDF);  __bf16* WxX_bf= hbuf((size_t)CBG * CKA * CDF);
    float*  WpTm   = fbuf((size_t)CBG * CKA * SEQL); __bf16* WpT_bf= hbuf((size_t)CBG * CKA * SEQL);
    float*  H_c    = fbuf((size_t)CBG * CKA * CDF);
    float*  H_p    = fbuf((size_t)CBG * CKA * SEQL);
    float*  a_c    = fbuf((size_t)CBG * CDF);
    float*  a_p    = fbuf((size_t)CBG * SEQL);
    float*  cvec   = fbuf((size_t)CBG * CDH);
    float*  pvec   = fbuf((size_t)CBG * CPH);
    float*  xmean  = fbuf((size_t)CBG * CDH);
    float*  feat   = fbuf((size_t)CBG * 768);  __bf16* featn = hbuf((size_t)CBG * 768);
    float*  h1     = fbuf((size_t)CBG * 768);  __bf16* h1n   = hbuf((size_t)CBG * 768);
    float*  h2     = fbuf((size_t)CBG * CDH);

    // ---- launch helpers ----
    auto g1 = [](long long n) { return dim3((unsigned)((n + 255) / 256)); };
    auto cvt = [&](const float* s, __bf16* d, long long n) {
        k_cvt<<<g1(n), 256, 0, stream>>>(s, d, n);
    };
    auto cvtT = [&](const float* s, __bf16* d, int R, int C) {
        k_cvt_t<<<g1((long long)R * C), 256, 0, stream>>>(s, d, R, C);
    };
    auto fill0 = [&](float* p, long long n) {
        k_fill<<<g1(n), 256, 0, stream>>>(p, 0.f, n);
    };
    auto gemm = [&](const __bf16* A, const __bf16* B, float* C, const float* bias,
                    const float* add, int M, int Nn, int K, int act, int batch,
                    long long sA, long long sB, long long sC, long long sAdd) {
        dim3 grid((Nn + 63) / 64, (M + 15) / 16, batch);
        k_gemm_wmma<<<grid, 128, 0, stream>>>(A, B, C, bias, add, M, Nn, K, act,
                                              sA, sB, sC, sAdd);
    };
    const float* NOF = nullptr;

    // ---- weight prep (transpose f32 -> bf16 [N,K]) ----
    cvt (node_x, nodex_bf, (long long)GN * CDX);
    cvt (proteins, prot_bf, (long long)CBG * SEQL * CDP);
    cvtT(W0, W0T, CDX, CDH);      cvtT(dm_W, dmT, CDH, CDF);
    cvtT(Wu, WuT, CDF, CDF);      cvtT(Wv, WvT, CDF, CDF);
    cvtT(Wg, WgT, CDF, CDF);      cvtT(Wa, WaT, CDF, CDH);
    cvtT(Wp0, Wp0T, CDP, CPH);    cvtT(m_in_W, minT, CPH, CZ);
    cvtT(m_out_W, moutT, CDIN, CPH); cvtT(W_b, WbT, CPH, CDH);
    cvt (W_x, Wx_bf, (long long)CKA * CDH);
    cvt (W_p, Wp_bf, (long long)CKA * CPH);
    cvtT(W1, W1T, 768, 768);      cvtT(W2, W2T, 768, CDH);

    // ---- drug branch ----
    gemm(nodex_bf, W0T, xd_pre, b0, NOF, GN, CDH, CDX, 1, 1, 0, 0, 0, 0);
    cvt (xd_pre, xdp_bf, (long long)GN * CDH);
    gemm(xdp_bf, dmT, x_in, dm_b, NOF, GN, CDF, CDH, 1, 1, 0, 0, 0, 0);
    cvt (x_in, xin_bf, (long long)GN * CDF);
    gemm(xin_bf, WuT, eu, NOF, NOF, GN, CDF, CDF, 0, 1, 0, 0, 0, 0);
    gemm(xin_bf, WvT, ev, NOF, NOF, GN, CDF, CDF, 0, 1, 0, 0, 0, 0);
    k_edge_init<<<GE, CDF, 0, stream>>>(eu, ev, edge_attr, We, eidx, e_attr);

    for (int it = 0; it < 3; ++it) {
        const float* prev = (it == 0) ? e_attr : outs[it - 1];
        fill0(segA, (long long)GE * CDF);
        k_scatter_add<<<g1((long long)GELG * CDF), 256, 0, stream>>>(prev, lgidx, lgidx + GELG, segA, GELG);
        k_add2<<<g1((long long)GE * CDF), 256, 0, stream>>>(e_attr, segA, outs[it], (long long)GE * CDF);
        fill0(segB, (long long)GE * CDF);
        k_scatter_add<<<g1((long long)GELG * CDF), 256, 0, stream>>>(outs[it], lgidx, lgidx + GELG, segB, GELG);
        k_xconv<<<g1(GE), 256, 0, stream>>>(segB, outs[it], attWrel, attbrel, attWroot, xconv);
        k_segsoftmax<<<CBG, CEPG, 0, stream>>>(xconv, scv);
        k_gx<<<CBG, CDF, 0, stream>>>(outs[it], scv, gx);
        cvt (gx, gx_bf, (long long)CBG * CDF);
        gemm(gx_bf, WgT, gout[it], bg, NOF, CBG, CDF, CDF, 2, 1, 0, 0, 0, 0);
    }
    k_scores<<<CBG, CDF, 0, stream>>>(gout[0], gout[1], gout[2], w_att, scores);
    k_mixout<<<g1((long long)GE * CDF), 256, 0, stream>>>(outs[0], outs[1], outs[2], scores, ebatch, outF);
    k_copy<<<g1((long long)GN * CDF), 256, 0, stream>>>(x_in, dn, (long long)GN * CDF);
    k_scatter_add<<<g1((long long)GE * CDF), 256, 0, stream>>>(outF, (const int*)nullptr, eidx + GE, dn, GE);
    cvt (dn, dn_bf, (long long)GN * CDF);
    gemm(dn_bf, WaT, encT, ba, NOF, GN, CDH, CDF, 1, 1, 0, 0, 0, 0);
    k_denc<<<g1((long long)GN * CDH), 256, 0, stream>>>(encT, xd_pre, denc, dd_bf);

    // ---- protein branch: target + mamba2 ----
    gemm(prot_bf, Wp0T, target, bp0, NOF, CBG * SEQL, CPH, CDP, 0, 1, 0, 0, 0, 0);
    cvt (target, tgt_bf, (long long)CBG * SEQL * CPH);
    gemm(tgt_bf, minT, zx, NOF, NOF, CBG * SEQL, CZ, CPH, 0, 1, 0, 0, 0, 0);
    k_dtda<<<g1((long long)CBG * SEQL * CNH), 256, 0, stream>>>(zx, m_dt_b, m_A_log, dtb_, dAb);
    k_conv<<<g1((long long)CBG * SEQL * CDXBC), 256, 0, stream>>>(zx, m_conv_w, m_conv_b, xBC);
    k_scan<<<CBG * CNH, CDST, 0, stream>>>(xBC, dAb, dtb_, ys);
    k_ypost<<<CBG * SEQL, 256, 0, stream>>>(ys, xBC, zx, m_D, m_norm_w, y_bf);
    gemm(y_bf, moutT, mo, NOF, NOF, CBG * SEQL, CPH, CDIN, 0, 1, 0, 0, 0, 0);
    k_tmmean<<<g1((long long)CBG * CPH), 256, 0, stream>>>(mo, target, tmm);

    // ---- bilinear co-attention ----
    gemm(tgt_bf, WbT, T2, NOF, NOF, CBG * SEQL, CDH, CPH, 0, 1, 0, 0, 0, 0);
    cvt (T2, T2_bf, (long long)CBG * SEQL * CDH);
    gemm(T2_bf, dd_bf, Cc, NOF, NOF, SEQL, CDF, CDH, 2, CBG,
         (long long)SEQL * CDH, (long long)CNPG * CDH, (long long)SEQL * CDF, 0);
    cvt (Cc, Cc_bf, (long long)CBG * SEQL * CDF);
    gemm(dd_bf, T2_bf, CcT, NOF, NOF, CDF, SEQL, CDH, 2, CBG,
         (long long)CNPG * CDH, (long long)SEQL * CDH, (long long)CDF * SEQL, 0);
    cvt (CcT, CcT_bf, (long long)CBG * CDF * SEQL);
    gemm(Wx_bf, dd_bf, WxX, NOF, NOF, CKA, CDF, CDH, 0, CBG,
         0, (long long)CNPG * CDH, (long long)CKA * CDF, 0);
    cvt (WxX, WxX_bf, (long long)CBG * CKA * CDF);
    gemm(Wp_bf, tgt_bf, WpTm, NOF, NOF, CKA, SEQL, CPH, 0, CBG,
         0, (long long)SEQL * CPH, (long long)CKA * SEQL, 0);
    cvt (WpTm, WpT_bf, (long long)CBG * CKA * SEQL);
    gemm(WpT_bf, CcT_bf, H_c, NOF, WxX, CKA, CDF, SEQL, 2, CBG,
         (long long)CKA * SEQL, (long long)CDF * SEQL, (long long)CKA * CDF, (long long)CKA * CDF);
    gemm(WxX_bf, Cc_bf, H_p, NOF, WpTm, CKA, SEQL, CDF, 2, CBG,
         (long long)CKA * CDF, (long long)SEQL * CDF, (long long)CKA * SEQL, (long long)CKA * SEQL);
    k_attn<<<CBG, CDF, CDF * sizeof(float), stream>>>(H_c, w_hx, a_c, CDF);
    k_attn<<<CBG, SEQL, SEQL * sizeof(float), stream>>>(H_p, w_hp, a_p, SEQL);
    k_wsum<<<CBG, CDH, 0, stream>>>(a_c, denc, cvec, CNPG, CDH);
    k_wsum<<<CBG, CPH, 0, stream>>>(a_p, target, pvec, SEQL, CPH);
    k_xmean<<<CBG, CDH, 0, stream>>>(xd_pre, xmean);

    // ---- head ----
    k_feat<<<g1((long long)CBG * 768), 256, 0, stream>>>(xmean, cvec, tmm, pvec, feat);
    k_ln<<<CBG, 256, 0, stream>>>(feat, ln1_g, ln1_b, featn, 768);
    gemm(featn, W1T, h1, b1, NOF, CBG, 768, 768, 1, 1, 0, 0, 0, 0);
    k_ln<<<CBG, 256, 0, stream>>>(h1, ln2_g, ln2_b, h1n, 768);
    gemm(h1n, W2T, h2, b2, NOF, CBG, CDH, 768, 1, 1, 0, 0, 0, 0);
    k_final<<<1, 32, 0, stream>>>(h2, W3, b3, (float*)d_out);
}
